// PropertyFocusedAttention_18657337933876
// MI455X (gfx1250) — matrix operand order repaired and checked by reading, not compiled, not run on hardware
//
#include <hip/hip_runtime.h>

typedef __attribute__((ext_vector_type(16))) __bf16 v16bf;
typedef __attribute__((ext_vector_type(8)))  float  v8f;
typedef __attribute__((ext_vector_type(4)))  unsigned int u32x4;
typedef __attribute__((ext_vector_type(8)))  unsigned int u32x8;

#define BATCH    65536
#define SEQ      5
#define DMODEL   128
#define HEADS    8
#define HEADDIM  16
#define NB       16                  // batches per workgroup
#define ROWS     (NB * SEQ)          // 80 rows = 5 M-tiles of 16
#define NTHREADS 256                 // 8 wave32
#define WELEMS   (4 * DMODEL * DMODEL)   // 65536 bf16 weight elements

__device__ __forceinline__ unsigned short f2bfr(float f) {
  unsigned u = __builtin_bit_cast(unsigned, f);
  u += 0x7FFFu + ((u >> 16) & 1u);          // round-to-nearest-even
  return (unsigned short)(u >> 16);
}

// A fragment: 16x32 bf16, rows in lanes (lane&15), K halves split by lane half.
__device__ __forceinline__ v16bf frag_a(const unsigned short* base, int mBase,
                                        int kk, int l15, int lhalf) {
  const unsigned short* p = base + (mBase + l15) * DMODEL + kk + lhalf * 16;
  v16bf a;
#pragma unroll
  for (int e = 0; e < 8; ++e) a[e] = __builtin_bit_cast(__bf16, p[e]);
#pragma unroll
  for (int e = 0; e < 8; ++e) a[8 + e] = __builtin_bit_cast(__bf16, p[8 + e]);
  return a;
}

// B fragment: 32x16 bf16, columns in lanes, K halves split by lane half.
// Weights staged TRANSPOSED ([n][k]) so K is contiguous.
__device__ __forceinline__ v16bf frag_b(const unsigned short* wT, int nBase,
                                        int kk, int l15, int lhalf) {
  const unsigned short* p = wT + (nBase + l15) * DMODEL + kk + lhalf * 16;
  v16bf b;
#pragma unroll
  for (int e = 0; e < 16; ++e) b[e] = __builtin_bit_cast(__bf16, p[e]);
  return b;
}

// ---- prep: one-time fp32 -> bf16 transposed weight staging into d_ws.
// Output layout matches LDS exactly: wbf[w*16384 + n*128 + k] = bf16(W_w[k][n]).
__global__ __launch_bounds__(NTHREADS) void pfa_prep_weights(
    const float* __restrict__ Wq, const float* __restrict__ Wk,
    const float* __restrict__ Wv, const float* __restrict__ Wo,
    unsigned short* __restrict__ wbf) {
  int i = blockIdx.x * NTHREADS + threadIdx.x;   // 0..65535, output-ordered
  const float* wsrc[4] = {Wq, Wk, Wv, Wo};
  int w = i >> 14, r = i & 16383;
  int n = r >> 7, k = r & 127;
  wbf[i] = f2bfr(wsrc[w][k * DMODEL + n]);
}

__global__ __launch_bounds__(NTHREADS) void pfa_fused_kernel(
    const float* __restrict__ x,
    const unsigned short* __restrict__ wbf,      // pre-staged bf16 weights (128KB)
    const float* __restrict__ bq, const float* __restrict__ bk,
    const float* __restrict__ bv, const float* __restrict__ bo,
    const float* __restrict__ pbias,
    float* __restrict__ out) {
  __shared__ __attribute__((aligned(128))) unsigned short sW[WELEMS];         // 128KB
  __shared__ __attribute__((aligned(16)))  unsigned short sX[ROWS * DMODEL];  // 20KB
  __shared__ __attribute__((aligned(16)))  float sQKV[3 * ROWS * DMODEL];     // 120KB fp32
  __shared__ float sPB[HEADS * SEQ * SEQ];
  __shared__ float sBias[4 * DMODEL];

  const int tid   = threadIdx.x;
  const int lane  = tid & 31;
  const int wave  = tid >> 5;
  const int lhalf = (lane >> 4) & 1;
  const int l15   = lane & 15;

  // ---- stage 0a: one TDM DMA pulls all four bf16 weight matrices into LDS.
  if (wave == 0) {
    unsigned lds_addr = (unsigned)(unsigned long long)&sW[0];
    unsigned long long ga = (unsigned long long)wbf;
    u32x4 g0;
    g0[0] = 1u;                                        // count=1, user descriptor
    g0[1] = lds_addr;                                  // lds_addr [63:32]
    g0[2] = (unsigned)(ga & 0xFFFFFFFFull);            // global_addr [95:64]
    g0[3] = (unsigned)((ga >> 32) & 0x01FFFFFFull)     // global_addr [120:96]
            | (2u << 30);                              // type=2 ("image") [127:126]
    u32x8 g1;
    g1[0] = 2u << 16;          // workgroup_mask=0, data_size=2 (4-byte elements)
    g1[1] = 0x8000u << 16;     // tensor_dim0 = 32768 (low 16 bits at [63:48])
    g1[2] = 1u << 16;          // tensor_dim0 hi=0, tensor_dim1 = 1
    g1[3] = 0x8000u << 16;     // tensor_dim1 hi=0, tile_dim0 = 32768
    g1[4] = 1u;                // tile_dim1 = 1, tile_dim2 = 0
    g1[5] = 32768u;            // tensor_dim0_stride = 32768
    g1[6] = 0u;
    g1[7] = 0u;
    asm volatile("tensor_load_to_lds %0, %1" :: "s"(g0), "s"(g1) : "memory");
  }

  // ---- stage 0b: convert this block's x tile fp32 -> packed bf16 (overlaps DMA)
  const long long xbase = (long long)blockIdx.x * (ROWS * DMODEL);
  {
    const float4* xv = (const float4*)(x + xbase);
    unsigned* sX32 = (unsigned*)sX;
#pragma unroll 1
    for (int i = tid; i < ROWS * DMODEL / 4; i += NTHREADS) {
      float4 f = xv[i];
      sX32[2 * i + 0] = ((unsigned)f2bfr(f.y) << 16) | f2bfr(f.x);
      sX32[2 * i + 1] = ((unsigned)f2bfr(f.w) << 16) | f2bfr(f.z);
    }
  }
  for (int i = tid; i < HEADS * SEQ * SEQ; i += NTHREADS) sPB[i] = pbias[i];
  {
    const float* bsrc[4] = {bq, bk, bv, bo};
    for (int i = tid; i < 4 * DMODEL; i += NTHREADS) sBias[i] = bsrc[i >> 7][i & 127];
  }
  if (wave == 0) __builtin_amdgcn_s_wait_tensorcnt(0);
  __syncthreads();

  // ---- stage 1: QKV projections. 24 column tiles (3 mats x 8 N-tiles);
  // each wave owns 3 column tiles, hoists B fragments, sweeps 5 M-tiles.
#pragma unroll 1
  for (int ct = wave; ct < 24; ct += 8) {
    int mat = ct >> 3, nt = ct & 7;
    const unsigned short* wT = sW + mat * 16384;
    const int nB = nt * 16;
    v16bf b0 = frag_b(wT, nB,  0, l15, lhalf);
    v16bf b1 = frag_b(wT, nB, 32, l15, lhalf);
    v16bf b2 = frag_b(wT, nB, 64, l15, lhalf);
    v16bf b3 = frag_b(wT, nB, 96, l15, lhalf);
    int n = nB + l15;
    float bias = sBias[mat * DMODEL + n];
    float* dst = sQKV + mat * (ROWS * DMODEL);
#pragma unroll 1
    for (int mt = 0; mt < 5; ++mt) {
      v8f c = {};
      c = __builtin_amdgcn_wmma_f32_16x16x32_bf16(false, frag_a(sX, mt*16,  0, l15, lhalf),
                                                  false, b0, (short)0, c, false, false);
      c = __builtin_amdgcn_wmma_f32_16x16x32_bf16(false, frag_a(sX, mt*16, 32, l15, lhalf),
                                                  false, b1, (short)0, c, false, false);
      c = __builtin_amdgcn_wmma_f32_16x16x32_bf16(false, frag_a(sX, mt*16, 64, l15, lhalf),
                                                  false, b2, (short)0, c, false, false);
      c = __builtin_amdgcn_wmma_f32_16x16x32_bf16(false, frag_a(sX, mt*16, 96, l15, lhalf),
                                                  false, b3, (short)0, c, false, false);
#pragma unroll
      for (int jj = 0; jj < 8; ++jj) {                  // C/D: VGPR j -> row j+8*half
        int m = mt * 16 + lhalf * 8 + jj;
        dst[m * DMODEL + n] = c[jj] + bias;             // fp32, no conversion
      }
    }
  }
  __syncthreads();

  // ---- stage 2: attention; one (batch, head, query) item per thread pass.
#pragma unroll 1
  for (int it = tid; it < NB * HEADS * SEQ; it += NTHREADS) {   // 640 items
    int s = it % SEQ;
    int bh = it / SEQ;
    int h = bh & 7, bl = bh >> 3;
    int rowB = bl * SEQ, cb = h * HEADDIM;
    const float* Qs = sQKV + (rowB + s) * DMODEL + cb;
    const float* Ks = sQKV + ROWS * DMODEL + rowB * DMODEL + cb;
    const float* Vs = sQKV + 2 * ROWS * DMODEL + rowB * DMODEL + cb;
    float q[HEADDIM];
#pragma unroll
    for (int d = 0; d < HEADDIM; ++d) q[d] = Qs[d];
    float sc[SEQ];
#pragma unroll
    for (int tt = 0; tt < SEQ; ++tt) {
      float acc = 0.f;
#pragma unroll
      for (int d = 0; d < HEADDIM; ++d) acc += q[d] * Ks[tt * DMODEL + d];
      sc[tt] = acc * 0.25f + sPB[h * 25 + s * 5 + tt];   // 1/sqrt(16)
    }
    float mx = sc[0];
#pragma unroll
    for (int tt = 1; tt < SEQ; ++tt) mx = fmaxf(mx, sc[tt]);
    float den = 0.f;
#pragma unroll
    for (int tt = 0; tt < SEQ; ++tt) { sc[tt] = __expf(sc[tt] - mx); den += sc[tt]; }
    float inv = 1.0f / den;
#pragma unroll
    for (int d = 0; d < HEADDIM; ++d) {
      float acc = 0.f;
#pragma unroll
      for (int tt = 0; tt < SEQ; ++tt) acc += sc[tt] * Vs[tt * DMODEL + d];
      sX[(rowB + s) * DMODEL + cb + d] = f2bfr(acc * inv);  // context over x buffer
    }
  }
  __syncthreads();

  // ---- stage 3: output projection (context @ Wo + bo) -> global fp32.
  // 8 column tiles / 8 waves: hoist B fragments, sweep 5 M-tiles.
  {
    const unsigned short* wo = sW + 3 * 16384;
    int nt = wave;
    const int nB = nt * 16;
    v16bf b0 = frag_b(wo, nB,  0, l15, lhalf);
    v16bf b1 = frag_b(wo, nB, 32, l15, lhalf);
    v16bf b2 = frag_b(wo, nB, 64, l15, lhalf);
    v16bf b3 = frag_b(wo, nB, 96, l15, lhalf);
    int n = nB + l15;
    float bias = sBias[3 * DMODEL + n];
#pragma unroll 1
    for (int mt = 0; mt < 5; ++mt) {
      v8f c = {};
      c = __builtin_amdgcn_wmma_f32_16x16x32_bf16(false, frag_a(sX, mt*16,  0, l15, lhalf),
                                                  false, b0, (short)0, c, false, false);
      c = __builtin_amdgcn_wmma_f32_16x16x32_bf16(false, frag_a(sX, mt*16, 32, l15, lhalf),
                                                  false, b1, (short)0, c, false, false);
      c = __builtin_amdgcn_wmma_f32_16x16x32_bf16(false, frag_a(sX, mt*16, 64, l15, lhalf),
                                                  false, b2, (short)0, c, false, false);
      c = __builtin_amdgcn_wmma_f32_16x16x32_bf16(false, frag_a(sX, mt*16, 96, l15, lhalf),
                                                  false, b3, (short)0, c, false, false);
#pragma unroll
      for (int jj = 0; jj < 8; ++jj) {
        int m = mt * 16 + lhalf * 8 + jj;
        out[xbase + (long long)m * DMODEL + n] = c[jj] + bias;
      }
    }
  }
}

extern "C" void kernel_launch(void* const* d_in, const int* in_sizes, int n_in,
                              void* d_out, int out_size, void* d_ws, size_t ws_size,
                              hipStream_t stream) {
  const float* x     = (const float*)d_in[0];
  const float* Wq    = (const float*)d_in[1];
  const float* bq    = (const float*)d_in[2];
  const float* Wk    = (const float*)d_in[3];
  const float* bk    = (const float*)d_in[4];
  const float* Wv    = (const float*)d_in[5];
  const float* bv    = (const float*)d_in[6];
  const float* Wo    = (const float*)d_in[7];
  const float* bo    = (const float*)d_in[8];
  const float* pbias = (const float*)d_in[9];
  float* out = (float*)d_out;
  unsigned short* wbf = (unsigned short*)d_ws;   // 128KB bf16 staged weights

  pfa_prep_weights<<<dim3(WELEMS / NTHREADS), dim3(NTHREADS), 0, stream>>>(
      Wq, Wk, Wv, Wo, wbf);
  pfa_fused_kernel<<<dim3(BATCH / NB), dim3(NTHREADS), 0, stream>>>(
      x, wbf, bq, bk, bv, bo, pbias, out);
}
